// Attention4DDownsample_12987981103353
// MI455X (gfx1250) — matrix operand order repaired and checked by reading, not compiled, not run on hardware
//
#include <hip/hip_runtime.h>
#include <math.h>

typedef unsigned short u16;
typedef unsigned int   u32;
typedef __attribute__((ext_vector_type(16))) __bf16 v16bf;
typedef __attribute__((ext_vector_type(8)))  float  v8f;

struct Frag { union { u32 u[8]; v16bf v; }; };

__device__ inline u16 f2bf(float f) {
  union { float f; u32 u; } v; v.f = f;
  u32 r = v.u + 0x7fffu + ((v.u >> 16) & 1u);
  return (u16)(r >> 16);
}
__device__ inline float bf2f(u16 h) {
  union { u32 u; float f; } v; v.u = ((u32)h) << 16;
  return v.f;
}

// ---------------------------------------------------------------------------
// DPP 16-lane butterfly reductions (VALU only; no LDS round trips).
// 0xB1 = quad_perm[1,0,3,2] (xor1), 0x4E = quad_perm[2,3,0,1] (xor2),
// 0x141 = row_half_mirror (xor7 in 8), 0x140 = row_mirror (xor15 in 16).
// ---------------------------------------------------------------------------
template <int CTRL>
__device__ inline float dpp_mov(float v) {
  union { float f; int i; } a, b;
  a.f = v;
  b.i = __builtin_amdgcn_update_dpp(0, a.i, CTRL, 0xF, 0xF, true);
  return b.f;
}
__device__ inline float red_max16(float v) {
  v = fmaxf(v, dpp_mov<0xB1>(v));
  v = fmaxf(v, dpp_mov<0x4E>(v));
  v = fmaxf(v, dpp_mov<0x141>(v));
  v = fmaxf(v, dpp_mov<0x140>(v));
  return v;
}
__device__ inline float red_sum16(float v) {
  v += dpp_mov<0xB1>(v);
  v += dpp_mov<0x4E>(v);
  v += dpp_mov<0x141>(v);
  v += dpp_mov<0x140>(v);
  return v;
}

// ---------------------------------------------------------------------------
// f32 -> bf16 convert
// ---------------------------------------------------------------------------
__global__ void cvt_kernel(const float* __restrict__ in, u16* __restrict__ out, int n) {
  int i = blockIdx.x * 256 + threadIdx.x;
  if (i < n) out[i] = f2bf(in[i]);
}

// ---------------------------------------------------------------------------
// bias precompute: bias[h][196][784] = ab[h][bias_idxs[i]]
// ---------------------------------------------------------------------------
__global__ void bias_kernel(const float* __restrict__ ab, const int* __restrict__ idxs,
                            float* __restrict__ out, int n_off) {
  int i = blockIdx.x * 256 + threadIdx.x;
  if (i >= 196 * 784) return;
  int idx = idxs[i];
  for (int h = 0; h < 8; h++)
    out[(size_t)h * 196 * 784 + i] = ab[h * n_off + idx];
}

// ---------------------------------------------------------------------------
// q-prep: qin[b][c][m] = dwconv3x3_s2(x)+ql_b + pool(x), bf16 out [B,384,196]
// ---------------------------------------------------------------------------
__global__ void qprep_kernel(const float* __restrict__ x, const float* __restrict__ w,
                             const float* __restrict__ bias, u16* __restrict__ out) {
  int idx = blockIdx.x * 256 + threadIdx.x;
  if (idx >= 64 * 384 * 196) return;
  int m = idx % 196; int c = (idx / 196) % 384; int b = idx / (196 * 384);
  int i = m / 14, j = m % 14;
  const float* xb = x + ((size_t)b * 384 + c) * 784;
  float acc = bias[c];
  for (int di = 0; di < 3; di++) {
    int yy = 2 * i - 1 + di; if (yy < 0 || yy >= 28) continue;
    for (int dj = 0; dj < 3; dj++) {
      int xx = 2 * j - 1 + dj; if (xx < 0 || xx >= 28) continue;
      acc += w[c * 9 + di * 3 + dj] * xb[yy * 28 + xx];
    }
  }
  acc += xb[(2 * i) * 28 + 2 * j];   // pool_q
  out[idx] = f2bf(acc);
}

// ---------------------------------------------------------------------------
// v_local: bn(dwconv3x3_s2(v)+vl_b), f32 out [B,512,196]; v read as bf16
// ---------------------------------------------------------------------------
__global__ void vlocal_kernel(const u16* __restrict__ vbuf, const float* __restrict__ w,
                              const float* __restrict__ bias, const float* __restrict__ s,
                              const float* __restrict__ o, float* __restrict__ out) {
  int idx = blockIdx.x * 256 + threadIdx.x;
  if (idx >= 64 * 512 * 196) return;
  int m = idx % 196; int c = (idx / 196) % 512; int b = idx / (196 * 512);
  int i = m / 14, j = m % 14;
  const u16* vb = vbuf + ((size_t)b * 512 + c) * 784;
  float acc = bias[c];
  for (int di = 0; di < 3; di++) {
    int yy = 2 * i - 1 + di; if (yy < 0 || yy >= 28) continue;
    for (int dj = 0; dj < 3; dj++) {
      int xx = 2 * j - 1 + dj; if (xx < 0 || xx >= 28) continue;
      acc += w[c * 9 + di * 3 + dj] * bf2f(vb[yy * 28 + xx]);
    }
  }
  out[idx] = acc * s[c] + o[c];
}

// ---------------------------------------------------------------------------
// Tiled WMMA GEMM: C[M,N] = W[M,K]@X[K,N] per batch (blockIdx.z), fused BN.
// Block = 128 thr (4 waves); tile 64(M) x 64(N); K step 32.
// MODE 0: bf16 out [B,M,N]; MODE 1: bf16 q-layout [B,8,N,16]; MODE 2: f32 [B,M,N]
// ---------------------------------------------------------------------------
template <int MODE>
__global__ __launch_bounds__(128)
void gemm_bn(const u16* __restrict__ W, const u16* __restrict__ X,
             const float* __restrict__ cb, const float* __restrict__ cs,
             const float* __restrict__ co, void* __restrict__ out,
             int M, int K, int N) {
  __shared__ u16 Wt[64][32];
  __shared__ u16 Xt[64][32];   // transposed: Xt[n][k]
  const int b  = blockIdx.z;
  const int mb = blockIdx.y * 64;
  const int nb = blockIdx.x * 64;
  const int tid = threadIdx.x;
  const int wv = tid >> 5, ln = tid & 15, h2 = (tid >> 4) & 1;
  const u16* Xb = X + (size_t)b * K * N;

  v8f acc[4];
  for (int t = 0; t < 4; t++) for (int r = 0; r < 8; r++) acc[t][r] = 0.f;

  for (int kc = 0; kc < K; kc += 32) {
    for (int idx = tid; idx < 1024; idx += 128) {          // 64x32 W tile, u32 units
      int row = idx >> 4, col = (idx & 15) << 1;
      *(u32*)&Wt[row][col] = *(const u32*)&W[(size_t)(mb + row) * K + kc + col];
    }
    for (int e = tid; e < 2048; e += 128) {                // 32x64 X tile -> transposed
      int k = e >> 6, n = e & 63;
      u16 v = 0;
      if (nb + n < N) v = Xb[(size_t)(kc + k) * N + nb + n];
      Xt[n][k] = v;
    }
    __syncthreads();
    Frag a;
    for (int j = 0; j < 8; j++) {
      int k0 = ((j >> 2) << 4) + (h2 << 3) + ((j & 3) << 1);   // ISA 16-bit A layout
      a.u[j] = *(const u32*)&Wt[wv * 16 + ln][k0];
    }
    Frag bm[4];
    for (int t = 0; t < 4; t++)                                // B: lane=col, K striped
      for (int j = 0; j < 8; j++)
        bm[t].u[j] = *(const u32*)&Xt[t * 16 + ln][((j >> 2) << 4) + (h2 << 3) + ((j & 3) << 1)];
    for (int t = 0; t < 4; t++)                                // back-to-back WMMAs
      acc[t] = __builtin_amdgcn_wmma_f32_16x16x32_bf16(
          false, a.v, false, bm[t].v, (short)0, acc[t], false, false);
    __syncthreads();
  }
  for (int t = 0; t < 4; t++) {
    int gn = nb + t * 16 + ln;
    for (int r = 0; r < 8; r++) {
      int m = mb + wv * 16 + r + (h2 << 3);
      float val = (acc[t][r] + cb[m]) * cs[m] + co[m];
      if (gn < N) {
        if (MODE == 0) {
          ((u16*)out)[(size_t)(b * M + m) * N + gn] = f2bf(val);
        } else if (MODE == 1) {
          int hh = m >> 4, c = m & 15;
          ((u16*)out)[(((size_t)b * 8 + hh) * N + gn) * 16 + c] = f2bf(val);
        } else {
          ((float*)out)[(size_t)(b * M + m) * N + gn] = val;
        }
      }
    }
  }
}

// ---------------------------------------------------------------------------
// Fused attention: one workgroup per (b, head). 4 waves.
// LDS: kT[800][16], vT[64][800], q[208][16], Pstage[4][16][32]  (~135.5 KB dyn)
// Flash softmax: online max/sum with DPP half-wave butterflies.
// All inner-loop control is branchless (zero-row redirect + clamps) so EXEC
// stays all-ones through the WMMA stream.
// Output = gelu(O/l + v_local) as bf16 -> gbuf [B,512,196]
// ---------------------------------------------------------------------------
__global__ __launch_bounds__(128)
void attn_kernel(const u16* __restrict__ qbuf, const u16* __restrict__ kbuf,
                 const u16* __restrict__ vbuf, const float* __restrict__ biasb,
                 const float* __restrict__ vlocal, u16* __restrict__ gbuf) {
  extern __shared__ u16 smem[];
  u16* kl = smem;                  // [800][16]  rows 784..799 zeroed
  u16* vt = kl + 800 * 16;         // [64][800]
  u16* ql = vt + 64 * 800;         // [208][16]
  u16* ps = ql + 208 * 16;         // [4][16][32]
  const int b = blockIdx.x >> 3, hd = blockIdx.x & 7;
  const int tid = threadIdx.x, wv = tid >> 5, ln = tid & 15, h2 = (tid >> 4) & 1;

  // stage k transposed: kl[n][kc]
  const u16* kg = kbuf + ((size_t)b * 128 + hd * 16) * 784;
  for (int i = tid; i < 16 * 784; i += 128) {
    int kc = i / 784, n = i - kc * 784;
    kl[n * 16 + kc] = kg[(size_t)kc * 784 + n];
  }
  for (int i = tid; i < 256; i += 128) kl[(784 + (i >> 4)) * 16 + (i & 15)] = 0;
  // stage v: vt[d][s], zero-pad s 784..799
  const u16* vg = vbuf + ((size_t)b * 512 + hd * 64) * 784;
  for (int i = tid; i < 64 * 400; i += 128) {
    int d = i / 400, sp = i - d * 400;
    u32 v = 0;
    if (sp < 392) v = *(const u32*)&vg[(size_t)d * 784 + sp * 2];
    *(u32*)&vt[d * 800 + sp * 2] = v;
  }
  // stage q [196][16] -> ql[208][16], pad rows zero
  const u16* qg = qbuf + ((size_t)b * 8 + hd) * 196 * 16;
  for (int i = tid; i < 208 * 8; i += 128) {
    u32 v = 0;
    if (i < 196 * 8) v = *(const u32*)&qg[i * 2];
    *(u32*)&ql[i * 2] = v;
  }
  __syncthreads();

  const float* biash = biasb + (size_t)hd * 196 * 784;
  for (int chunk = wv; chunk < 13; chunk += 4) {
    int mb = chunk * 16;
    Frag qa;                                  // A-frag: 16 rows x K32 (K>=16 zero)
    for (int j = 0; j < 8; j++) {
      int k0 = (h2 << 4) + (j << 1);
      qa.u[j] = (k0 < 16) ? *(const u32*)&ql[(mb + ln) * 16 + k0] : 0u;
    }
    v8f o[4]; float mrow[8], lrow[8], s0[8], s1[8];
    for (int d = 0; d < 4; d++) for (int r = 0; r < 8; r++) o[d][r] = 0.f;
    for (int r = 0; r < 8; r++) { mrow[r] = -1e30f; lrow[r] = 0.f; }

    // per-row clamped bias base (pad rows reuse row 195; results discarded)
    int mc[8];
    for (int r = 0; r < 8; r++) {
      int m = mb + r + (h2 << 3);
      mc[r] = m < 196 ? m : 195;
    }

    for (int kk = 0; kk < 800; kk += 32) {
      // two 16-col S tiles. Lanes 16-31 (K16-31 = zero pad) read the zeroed
      // pad rows 784..799 -> unconditional ds_loads, no exec divergence.
      int row0 = h2 ? (784 + ln) : (kk + ln);
      int row1 = h2 ? (784 + ln) : (kk + 16 + ln);
      Frag kf0, kf1;
      for (int j = 0; j < 8; j++) {
        kf0.u[j] = *(const u32*)&kl[row0 * 16 + (j << 1)];
        kf1.u[j] = *(const u32*)&kl[row1 * 16 + (j << 1)];
      }
      v8f c0, c1;
      for (int r = 0; r < 8; r++) { c0[r] = 0.f; c1[r] = 0.f; }
      c0 = __builtin_amdgcn_wmma_f32_16x16x32_bf16(
          false, qa.v, false, kf0.v, (short)0, c0, false, false);
      c1 = __builtin_amdgcn_wmma_f32_16x16x32_bf16(
          false, qa.v, false, kf1.v, (short)0, c1, false, false);
      {
        int col0 = kk + ln;                    // always < 784
        int col1 = kk + 16 + ln;
        int col1c = col1 < 784 ? col1 : 783;
        bool tail = (kk + 16) >= 784;          // wave-uniform (last iter only)
        for (int r = 0; r < 8; r++) {
          s0[r] = c0[r] * 0.25f + biash[(size_t)mc[r] * 784 + col0];
          float sv = c1[r] * 0.25f + biash[(size_t)mc[r] * 784 + col1c];
          s1[r] = tail ? -1e30f : sv;
        }
      }
      // online softmax update (DPP half-wave butterflies)
      for (int r = 0; r < 8; r++) {
        float tm = red_max16(fmaxf(s0[r], s1[r]));
        float nm = fmaxf(mrow[r], tm);
        float corr = __expf(mrow[r] - nm);
        mrow[r] = nm;
        float p0 = __expf(s0[r] - nm), p1 = __expf(s1[r] - nm);
        s0[r] = p0; s1[r] = p1;
        float sum = red_sum16(p0 + p1);
        lrow[r] = lrow[r] * corr + sum;
        for (int d = 0; d < 4; d++) o[d][r] *= corr;
      }
      // stage P (16x32 bf16) per wave, then use as A-frag for P@V
      u16* pw = ps + wv * 512;
      for (int r = 0; r < 8; r++) {
        int m = r + (h2 << 3);
        pw[m * 32 + ln]      = f2bf(s0[r]);
        pw[m * 32 + 16 + ln] = f2bf(s1[r]);
      }
      asm volatile("s_wait_dscnt 0" ::: "memory");   // intra-wave LDS RAW
      Frag pa;
      for (int j = 0; j < 8; j++) {
        int k0 = (h2 << 4) + (j << 1);
        pa.u[j] = *(const u32*)&pw[ln * 32 + k0];
      }
      Frag vf[4];
      for (int d = 0; d < 4; d++)
        for (int j = 0; j < 8; j++)
          vf[d].u[j] = *(const u32*)&vt[(d * 16 + ln) * 800 + kk + (h2 << 4) + (j << 1)];
      for (int d = 0; d < 4; d++)                    // back-to-back WMMAs
        o[d] = __builtin_amdgcn_wmma_f32_16x16x32_bf16(
            false, pa.v, false, vf[d].v, (short)0, o[d], false, false);
    }
    // epilogue: normalize, + v_local, gelu(exact), store bf16
    for (int d = 0; d < 4; d++) {
      int ch = hd * 64 + d * 16 + ln;
      for (int r = 0; r < 8; r++) {
        int m = mb + r + (h2 << 3);
        if (m < 196) {
          size_t oi = (size_t)(b * 512 + ch) * 196 + m;
          float val = o[d][r] / lrow[r] + vlocal[oi];
          float g = 0.5f * val * (1.f + erff(val * 0.70710678118654752f));
          gbuf[oi] = f2bf(g);
        }
      }
    }
  }
}

// ---------------------------------------------------------------------------
extern "C" void kernel_launch(void* const* d_in, const int* in_sizes, int n_in,
                              void* d_out, int out_size, void* d_ws, size_t ws_size,
                              hipStream_t stream) {
  const float* x    = (const float*)d_in[0];
  const float* ql_w = (const float*)d_in[1];
  const float* ql_b = (const float*)d_in[2];
  const float* qp_w = (const float*)d_in[3];
  const float* qp_b = (const float*)d_in[4];
  const float* qp_s = (const float*)d_in[5];
  const float* qp_o = (const float*)d_in[6];
  const float* k_w  = (const float*)d_in[7];
  const float* k_b  = (const float*)d_in[8];
  const float* k_s  = (const float*)d_in[9];
  const float* k_o  = (const float*)d_in[10];
  const float* v_w  = (const float*)d_in[11];
  const float* v_b  = (const float*)d_in[12];
  const float* v_s  = (const float*)d_in[13];
  const float* v_o  = (const float*)d_in[14];
  const float* vl_w = (const float*)d_in[15];
  const float* vl_b = (const float*)d_in[16];
  const float* vl_s = (const float*)d_in[17];
  const float* vl_o = (const float*)d_in[18];
  const float* p_w  = (const float*)d_in[19];
  const float* p_b  = (const float*)d_in[20];
  const float* p_s  = (const float*)d_in[21];
  const float* p_o  = (const float*)d_in[22];
  const float* ab   = (const float*)d_in[23];
  const int*   bidx = (const int*)d_in[24];
  const int n_off = in_sizes[23] / 8;

  char* ws = (char*)d_ws;
  size_t off = 0;
  auto alloc = [&](size_t bytes) -> char* {
    char* p = ws + off;
    off = (off + bytes + 255) & ~(size_t)255;
    return p;
  };
  u16*   xb    = (u16*)  alloc((size_t)64 * 384 * 784 * 2);
  u16*   wkb   = (u16*)  alloc((size_t)128 * 384 * 2);
  u16*   wvb   = (u16*)  alloc((size_t)512 * 384 * 2);
  u16*   wqb   = (u16*)  alloc((size_t)128 * 384 * 2);
  u16*   wpb   = (u16*)  alloc((size_t)384 * 512 * 2);
  u16*   kbuf  = (u16*)  alloc((size_t)64 * 128 * 784 * 2);
  u16*   vbuf  = (u16*)  alloc((size_t)64 * 512 * 784 * 2);
  u16*   qin   = (u16*)  alloc((size_t)64 * 384 * 196 * 2);
  u16*   qbuf  = (u16*)  alloc((size_t)64 * 8 * 196 * 16 * 2);
  float* biasb = (float*)alloc((size_t)8 * 196 * 784 * 4);
  float* vloc  = (float*)alloc((size_t)64 * 512 * 196 * 4);
  u16*   gbuf  = (u16*)  alloc((size_t)64 * 512 * 196 * 2);

  auto cvt = [&](const float* in, u16* out, int n) {
    cvt_kernel<<<dim3((n + 255) / 256), dim3(256), 0, stream>>>(in, out, n);
  };
  cvt(x,    xb,  64 * 384 * 784);
  cvt(k_w,  wkb, 128 * 384);
  cvt(v_w,  wvb, 512 * 384);
  cvt(qp_w, wqb, 128 * 384);
  cvt(p_w,  wpb, 384 * 512);

  bias_kernel<<<dim3((196 * 784 + 255) / 256), dim3(256), 0, stream>>>(ab, bidx, biasb, n_off);

  // k = bn(k_w @ x):   [B,128,784]
  gemm_bn<0><<<dim3(13, 2, 64), dim3(128), 0, stream>>>(wkb, xb, k_b, k_s, k_o, kbuf, 128, 384, 784);
  // v = bn(v_w @ x):   [B,512,784]
  gemm_bn<0><<<dim3(13, 8, 64), dim3(128), 0, stream>>>(wvb, xb, v_b, v_s, v_o, vbuf, 512, 384, 784);

  // q input = dwconv(x)+bias + pool(x)  -> bf16 [B,384,196]
  qprep_kernel<<<dim3((64 * 384 * 196 + 255) / 256), dim3(256), 0, stream>>>(x, ql_w, ql_b, qin);
  // q = bn(qp_w @ qin): stored [B,8,196,16]
  gemm_bn<1><<<dim3(4, 2, 64), dim3(128), 0, stream>>>(wqb, qin, qp_b, qp_s, qp_o, qbuf, 128, 384, 196);

  // v_local = bn(dwconv(v)+vl_b)  -> f32 [B,512,196]
  vlocal_kernel<<<dim3((64 * 512 * 196 + 255) / 256), dim3(256), 0, stream>>>(vbuf, vl_w, vl_b, vl_s, vl_o, vloc);

  // fused flash attention + gelu(out + v_local) -> bf16 gbuf [B,512,196]
  const size_t attn_lds = (size_t)(800 * 16 + 64 * 800 + 208 * 16 + 4 * 16 * 32) * 2;
  attn_kernel<<<dim3(64 * 8), dim3(128), attn_lds, stream>>>(qbuf, kbuf, vbuf, biasb, vloc, gbuf);

  // out = bn(p_w @ gbuf): f32 [B,384,196] -> d_out
  gemm_bn<2><<<dim3(4, 6, 64), dim3(128), 0, stream>>>(wpb, gbuf, p_b, p_s, p_o, d_out, 384, 512, 196);
}
